// MSWIDTBlock_6828998001251
// MI455X (gfx1250) — matrix operand order repaired and checked by reading, not compiled, not run on hardware
//
#include <hip/hip_runtime.h>
#include <math.h>

typedef __attribute__((ext_vector_type(16))) __bf16 bf16x16;
typedef __attribute__((ext_vector_type(8)))  __bf16 bf16x8;
typedef __attribute__((ext_vector_type(8)))  float  f32x8;
typedef __attribute__((ext_vector_type(4)))  float  f32x4;

static constexpr int Tn  = 4096;
static constexpr int Dn  = 512;
static constexpr int Hn  = 8;
static constexpr int HDn = 64;
static constexpr int FFn = 2048;

static __device__ __forceinline__ bf16x8 ld_bf8(const __bf16* p) {
  return *reinterpret_cast<const bf16x8*>(p);
}
static __device__ __forceinline__ bf16x16 cat8(bf16x8 lo, bf16x8 hi) {
  bf16x16 r;
#pragma unroll
  for (int i = 0; i < 8; ++i) { r[i] = lo[i]; r[i + 8] = hi[i]; }
  return r;
}
static __device__ __forceinline__ f32x8 wmma_bf16(bf16x16 a, bf16x16 b, f32x8 c) {
  return __builtin_amdgcn_wmma_f32_16x16x32_bf16(false, a, false, b, (short)0, c,
                                                 false, false);
}

// ---------------- weight convert + transpose: W[K,N] f32 -> WT[N,K] bf16 ---
__global__ void k_transpose_f32_to_bf16(const float* __restrict__ in,
                                        __bf16* __restrict__ out, int K, int N) {
  long id = (long)blockIdx.x * blockDim.x + threadIdx.x;
  long total = (long)K * N;
  if (id >= total) return;
  int k = (int)(id / N);
  int n = (int)(id % N);
  out[(long)n * K + k] = (__bf16)in[id];
}

// ---------------- layernorm (1 wave per row of 512) -> bf16 ----------------
// Each lane owns 16 contiguous elements: 4x b128 loads, 2x b128 bf16 stores.
__global__ void __launch_bounds__(128) k_layernorm_bf16(
    const float* __restrict__ x, const float* __restrict__ sc,
    const float* __restrict__ bi, __bf16* __restrict__ out) {
  int row  = blockIdx.x * 4 + (threadIdx.x >> 5);
  int lane = threadIdx.x & 31;
  const float* xr = x + (long)row * Dn + lane * 16;
  f32x4 v[4];
  float s = 0.f;
#pragma unroll
  for (int i = 0; i < 4; ++i) {
    v[i] = *reinterpret_cast<const f32x4*>(xr + i * 4);
#pragma unroll
    for (int e = 0; e < 4; ++e) s += v[i][e];
  }
#pragma unroll
  for (int o = 16; o >= 1; o >>= 1) s += __shfl_xor(s, o, 32);
  float mu = s * (1.0f / Dn);
  float q = 0.f;
#pragma unroll
  for (int i = 0; i < 4; ++i) {
#pragma unroll
    for (int e = 0; e < 4; ++e) { float d = v[i][e] - mu; q += d * d; }
  }
#pragma unroll
  for (int o = 16; o >= 1; o >>= 1) q += __shfl_xor(q, o, 32);
  float rs = rsqrtf(q * (1.0f / Dn) + 1e-6f);
  f32x4 scv[4], biv[4];
#pragma unroll
  for (int i = 0; i < 4; ++i) {
    scv[i] = *reinterpret_cast<const f32x4*>(sc + lane * 16 + i * 4);
    biv[i] = *reinterpret_cast<const f32x4*>(bi + lane * 16 + i * 4);
  }
  bf16x8 o0, o1;
#pragma unroll
  for (int i = 0; i < 4; ++i) {
#pragma unroll
    for (int e = 0; e < 4; ++e) {
      int idx = i * 4 + e;
      float val = (v[i][e] - mu) * rs * scv[i][e] + biv[i][e];
      if (idx < 8) o0[idx] = (__bf16)val;
      else         o1[idx - 8] = (__bf16)val;
    }
  }
  __bf16* op = out + (long)row * Dn + lane * 16;
  *reinterpret_cast<bf16x8*>(op)     = o0;
  *reinterpret_cast<bf16x8*>(op + 8) = o1;
}

// ---------------- generic bf16 WMMA GEMM: C = A[M,K] @ BT[N,K]^T + bias ----
// MODE 1: bf16 out, per-head [H][M][64] layout (Q/K)
// MODE 2: bf16 out, transposed [N][M] layout (V^T)
// MODE 3: f32 out, rowmajor, += residual
// MODE 4: bf16 out, rowmajor, tanh-GELU applied
template <int MODE>
__global__ void __launch_bounds__(128) k_gemm_wmma(
    const __bf16* __restrict__ A, const __bf16* __restrict__ BT,
    const float* __restrict__ bias, const float* __restrict__ res,
    void* __restrict__ outv, int M, int N, int K) {
  int w    = threadIdx.x >> 5;
  int lane = threadIdx.x & 31;
  int hv   = lane >> 4;
  int l16  = lane & 15;
  int m0 = (blockIdx.y * 4 + w) << 4;
  int n0 = blockIdx.x << 6;
  const __bf16* Ar = A + (long)(m0 + l16) * K;
  f32x8 acc[4] = {};
#pragma unroll 2
  for (int k0 = 0; k0 < K; k0 += 32) {
    int ka = k0 + hv * 8;
    bf16x16 af = cat8(ld_bf8(Ar + ka), ld_bf8(Ar + ka + 16));
#pragma unroll
    for (int j = 0; j < 4; ++j) {
      const __bf16* Bp = BT + (long)(n0 + j * 16 + l16) * K + k0 + hv * 8;
      bf16x16 bfm = cat8(ld_bf8(Bp), ld_bf8(Bp + 16));
      acc[j] = wmma_bf16(af, bfm, acc[j]);
    }
  }
#pragma unroll
  for (int j = 0; j < 4; ++j) {
#pragma unroll
    for (int r = 0; r < 8; ++r) {
      int m = m0 + r + hv * 8;
      int n = n0 + j * 16 + l16;
      float vv = acc[j][r] + bias[n];
      if (MODE == 1) {
        ((__bf16*)outv)[(long)(n >> 6) * M * 64 + (long)m * 64 + (n & 63)] =
            (__bf16)vv;
      } else if (MODE == 2) {
        ((__bf16*)outv)[(long)n * M + m] = (__bf16)vv;
      } else if (MODE == 3) {
        ((float*)outv)[(long)m * N + n] = vv + res[(long)m * N + n];
      } else if (MODE == 4) {
        float g = 0.5f * vv *
                  (1.0f + tanhf(0.7978845608f * (vv + 0.044715f * vv * vv * vv)));
        ((__bf16*)outv)[(long)m * N + n] = (__bf16)g;
      }
    }
  }
}

// ---------------- flash attention: wave = (head, 16 query rows) ------------
// 64-key tiles per iteration: 16 WMMAs per softmax/rescale pass.
// Qh/Kh: [H][T][64] bf16 ; VT: [H][64][T] bf16 ; O: [T][512] bf16
__global__ void __launch_bounds__(128) k_flash_attn(
    const __bf16* __restrict__ Qh, const __bf16* __restrict__ Kh,
    const __bf16* __restrict__ VT, __bf16* __restrict__ O) {
  __shared__ __bf16 plds[4][16 * 64];
  int w    = threadIdx.x >> 5;
  int lane = threadIdx.x & 31;
  int hv   = lane >> 4;
  int l16  = lane & 15;
  int gid = blockIdx.x * 4 + w;
  int h   = gid >> 8;   // Tn/16 == 256 query-blocks per head
  int qb  = gid & 255;
  const __bf16* Q  = Qh + (long)h * Tn * HDn;
  const __bf16* Kp = Kh + (long)h * Tn * HDn;
  const __bf16* Vp = VT + (long)h * HDn * Tn;
  int mrow = (qb << 4) + l16;
  // Q fragments for hd k-steps 0..31 / 32..63, pre-scaled by 1/sqrt(64)
  bf16x16 qf[2];
#pragma unroll
  for (int f = 0; f < 2; ++f) {
    int kb = f * 32 + hv * 8;
    bf16x16 t = cat8(ld_bf8(Q + (long)mrow * HDn + kb),
                     ld_bf8(Q + (long)mrow * HDn + kb + 16));
#pragma unroll
    for (int i = 0; i < 16; ++i) t[i] = (__bf16)((float)t[i] * 0.125f);
    qf[f] = t;
  }
  f32x8 o[4] = {};
  float mrun[8], lrun[8];
#pragma unroll
  for (int r = 0; r < 8; ++r) { mrun[r] = -INFINITY; lrun[r] = 0.f; }

  for (int kb = 0; kb < Tn; kb += 64) {
    if (kb + 64 < Tn) {  // pull next K/V tiles toward the WGP (global_prefetch_b8)
      __builtin_prefetch(Kp + (long)(kb + 64 + lane * 2) * HDn, 0, 1);
      __builtin_prefetch(Vp + (long)(lane * 2) * Tn + kb + 64, 0, 1);
    }
    // S = Q @ K^T for a 16x64 key tile (four 16-col WMMA tiles, two k-steps)
    f32x8 s[4] = {};
#pragma unroll
    for (int jn = 0; jn < 4; ++jn) {
      const __bf16* Kb = Kp + (long)(kb + jn * 16 + l16) * HDn;
#pragma unroll
      for (int ks = 0; ks < 2; ++ks) {
        int db = ks * 32 + hv * 8;
        bf16x16 kf = cat8(ld_bf8(Kb + db), ld_bf8(Kb + db + 16));
        s[jn] = wmma_bf16(qf[ks], kf, s[jn]);
      }
    }
    // online softmax per row; stash P tile (16x64 bf16) in LDS
#pragma unroll
    for (int r = 0; r < 8; ++r) {
      float s0 = s[0][r], s1 = s[1][r], s2 = s[2][r], s3 = s[3][r];
      float mx = fmaxf(fmaxf(s0, s1), fmaxf(s2, s3));
#pragma unroll
      for (int off = 8; off >= 1; off >>= 1)
        mx = fmaxf(mx, __shfl_xor(mx, off, 16));
      float mnew  = fmaxf(mrun[r], mx);
      float alpha = __expf(mrun[r] - mnew);
      float p0 = __expf(s0 - mnew);
      float p1 = __expf(s1 - mnew);
      float p2 = __expf(s2 - mnew);
      float p3 = __expf(s3 - mnew);
      float rs = (p0 + p1) + (p2 + p3);
#pragma unroll
      for (int off = 8; off >= 1; off >>= 1) rs += __shfl_xor(rs, off, 16);
      lrun[r] = lrun[r] * alpha + rs;
      mrun[r] = mnew;
      int prow = (r + hv * 8) * 64;
      plds[w][prow + l16]      = (__bf16)p0;
      plds[w][prow + 16 + l16] = (__bf16)p1;
      plds[w][prow + 32 + l16] = (__bf16)p2;
      plds[w][prow + 48 + l16] = (__bf16)p3;
#pragma unroll
      for (int j = 0; j < 4; ++j) o[j][r] *= alpha;
    }
    // re-stripe P (C-layout) into two A-fragments via LDS
    bf16x16 pa[2];
#pragma unroll
    for (int ks = 0; ks < 2; ++ks) {
      const __bf16* pp = &plds[w][l16 * 64 + ks * 32 + hv * 8];
      pa[ks] = cat8(ld_bf8(pp), ld_bf8(pp + 16));
    }
    // O += P @ V  (V columns contiguous in VT)
#pragma unroll
    for (int j = 0; j < 4; ++j) {
      const __bf16* Vb = Vp + (long)(j * 16 + l16) * Tn + kb + hv * 8;
#pragma unroll
      for (int ks = 0; ks < 2; ++ks) {
        bf16x16 vf = cat8(ld_bf8(Vb + ks * 32), ld_bf8(Vb + ks * 32 + 16));
        o[j] = wmma_bf16(pa[ks], vf, o[j]);
      }
    }
  }
  // normalize and store to [T][512]
#pragma unroll
  for (int r = 0; r < 8; ++r) {
    float inv = 1.0f / lrun[r];
    int row = (qb << 4) + r + hv * 8;
#pragma unroll
    for (int j = 0; j < 4; ++j) {
      O[(long)row * Dn + h * HDn + j * 16 + l16] = (__bf16)(o[j][r] * inv);
    }
  }
}

// ---------------------------------------------------------------------------
extern "C" void kernel_launch(void* const* d_in, const int* in_sizes, int n_in,
                              void* d_out, int out_size, void* d_ws,
                              size_t ws_size, hipStream_t stream) {
  (void)in_sizes; (void)n_in; (void)out_size; (void)ws_size;
  const float* x    = (const float*)d_in[0];
  const float* ln1s = (const float*)d_in[1];
  const float* ln1b = (const float*)d_in[2];
  const float* Wq   = (const float*)d_in[3];
  const float* bq   = (const float*)d_in[4];
  const float* Wk   = (const float*)d_in[5];
  const float* bk   = (const float*)d_in[6];
  const float* Wv   = (const float*)d_in[7];
  const float* bv   = (const float*)d_in[8];
  const float* Wo   = (const float*)d_in[9];
  const float* bo   = (const float*)d_in[10];
  const float* ln2s = (const float*)d_in[11];
  const float* ln2b = (const float*)d_in[12];
  const float* W1   = (const float*)d_in[13];
  const float* b1   = (const float*)d_in[14];
  const float* W2   = (const float*)d_in[15];
  const float* b2   = (const float*)d_in[16];
  float* out = (float*)d_out;

  char* p = (char*)d_ws;
  auto alloc = [&](size_t bytes) {
    char* r = p;
    p += (bytes + 255) & ~(size_t)255;
    return (void*)r;
  };
  __bf16* WqT = (__bf16*)alloc((size_t)Dn * Dn * 2);
  __bf16* WkT = (__bf16*)alloc((size_t)Dn * Dn * 2);
  __bf16* WvT = (__bf16*)alloc((size_t)Dn * Dn * 2);
  __bf16* WoT = (__bf16*)alloc((size_t)Dn * Dn * 2);
  __bf16* W1T = (__bf16*)alloc((size_t)Dn * FFn * 2);
  __bf16* W2T = (__bf16*)alloc((size_t)FFn * Dn * 2);
  __bf16* hb  = (__bf16*)alloc((size_t)Tn * Dn * 2);
  __bf16* Qh  = (__bf16*)alloc((size_t)Tn * Dn * 2);
  __bf16* Kh  = (__bf16*)alloc((size_t)Tn * Dn * 2);
  __bf16* Vt  = (__bf16*)alloc((size_t)Tn * Dn * 2);
  __bf16* Ob  = (__bf16*)alloc((size_t)Tn * Dn * 2);
  float*  x1  = (float*)alloc((size_t)Tn * Dn * 4);
  __bf16* h2b = (__bf16*)alloc((size_t)Tn * Dn * 2);
  __bf16* ffb = (__bf16*)alloc((size_t)Tn * FFn * 2);

  const int thr = 256;
  long we  = (long)Dn * Dn;
  long w1e = (long)Dn * FFn;
  k_transpose_f32_to_bf16<<<(we + thr - 1) / thr, thr, 0, stream>>>(Wq, WqT, Dn, Dn);
  k_transpose_f32_to_bf16<<<(we + thr - 1) / thr, thr, 0, stream>>>(Wk, WkT, Dn, Dn);
  k_transpose_f32_to_bf16<<<(we + thr - 1) / thr, thr, 0, stream>>>(Wv, WvT, Dn, Dn);
  k_transpose_f32_to_bf16<<<(we + thr - 1) / thr, thr, 0, stream>>>(Wo, WoT, Dn, Dn);
  k_transpose_f32_to_bf16<<<(w1e + thr - 1) / thr, thr, 0, stream>>>(W1, W1T, Dn, FFn);
  k_transpose_f32_to_bf16<<<(w1e + thr - 1) / thr, thr, 0, stream>>>(W2, W2T, FFn, Dn);

  k_layernorm_bf16<<<Tn / 4, 128, 0, stream>>>(x, ln1s, ln1b, hb);

  dim3 g512(Dn / 64, Tn / 64);
  k_gemm_wmma<1><<<g512, 128, 0, stream>>>(hb, WqT, bq, nullptr, Qh, Tn, Dn, Dn);
  k_gemm_wmma<1><<<g512, 128, 0, stream>>>(hb, WkT, bk, nullptr, Kh, Tn, Dn, Dn);
  k_gemm_wmma<2><<<g512, 128, 0, stream>>>(hb, WvT, bv, nullptr, Vt, Tn, Dn, Dn);

  k_flash_attn<<<(Hn * (Tn / 16)) / 4, 128, 0, stream>>>(Qh, Kh, Vt, Ob);

  k_gemm_wmma<3><<<g512, 128, 0, stream>>>(Ob, WoT, bo, x, x1, Tn, Dn, Dn);
  k_layernorm_bf16<<<Tn / 4, 128, 0, stream>>>(x1, ln2s, ln2b, h2b);

  dim3 gff(FFn / 64, Tn / 64);
  k_gemm_wmma<4><<<gff, 128, 0, stream>>>(h2b, W1T, b1, nullptr, ffb, Tn, FFn, Dn);
  k_gemm_wmma<3><<<g512, 128, 0, stream>>>(ffb, W2T, b2, x1, out, Tn, Dn, FFn);
}
